// ProteinStructureModel_86921548136927
// MI455X (gfx1250) — compile-verified, matrix operand order checked by reading, not run on hardware
//
#include <hip/hip_runtime.h>
#include <hip/hip_bf16.h>
#include <stdint.h>
#include <math.h>

typedef __attribute__((ext_vector_type(16))) _Float16 v16h;
typedef __attribute__((ext_vector_type(8)))  float    v8f;

#define LSEQ 512
#define NB 2
#define NPAIR (LSEQ*LSEQ)

__device__ inline float sigmoidf_(float x){ return 1.f/(1.f+__expf(-x)); }
__device__ inline float phi_elu(float t){ return t > 0.f ? t + 1.f : __expf(t); }

// ---- CDNA5 async global->LDS copy (ASYNCcnt path) ----
__device__ inline void async_b32(void* lds_dst, const void* gsrc){
  unsigned l = (unsigned)(uintptr_t)lds_dst;                  // low 32 bits of generic = LDS addr
  unsigned long long g = (unsigned long long)(uintptr_t)gsrc;
  asm volatile("global_load_async_to_lds_b32 %0, %1, off" :: "v"(l), "v"(g) : "memory");
}
__device__ inline void wait_async0(){ asm volatile("s_wait_asynccnt 0" ::: "memory"); }

// pack two contiguous float4s into 8 f16 halves of a v16h at [base, base+8)
__device__ inline void pack8lo(v16h& d, float4 x, float4 y){
  d[0]=(_Float16)x.x; d[1]=(_Float16)x.y; d[2]=(_Float16)x.z; d[3]=(_Float16)x.w;
  d[4]=(_Float16)y.x; d[5]=(_Float16)y.y; d[6]=(_Float16)y.z; d[7]=(_Float16)y.w;
}
__device__ inline void pack8hi(v16h& d, float4 x, float4 y){
  d[8]=(_Float16)x.x;  d[9]=(_Float16)x.y;  d[10]=(_Float16)x.z; d[11]=(_Float16)x.w;
  d[12]=(_Float16)y.x; d[13]=(_Float16)y.y; d[14]=(_Float16)y.z; d[15]=(_Float16)y.w;
}
// load one WMMA operand tile (16 halves) from a row pointer with unit k-stride
// (A row-major rows, or B^T rows): two contiguous 8-float runs at k0+8g, k0+16+8g.
__device__ inline v16h load_tile_rowmajor(const float* __restrict__ row, int k0, int g){
  const float4* p0 = (const float4*)(row + k0 + 8*g);
  const float4* p1 = (const float4*)(row + k0 + 16 + 8*g);
  v16h d;
  pack8lo(d, p0[0], p0[1]);
  pack8hi(d, p1[0], p1[1]);
  return d;
}

// =====================================================================
// Generic batched GEMM: C = A(MxK) x op(B) + bias, f16 WMMA, f32 accum.
// transB=1: B is (N,K) row-major (use B^T). transB=0: B is (K,N) row-major.
// One wave computes one 16x16 tile of C. Full K-tiles use b128 gathers;
// the K%32 tail uses clamped (branchless) gathers.
// =====================================================================
__global__ __launch_bounds__(32) void gemm_wmma_kernel(
    const float* __restrict__ A, int lda, long sA,
    const float* __restrict__ Bw, int ldb, long sB,
    const float* __restrict__ bias,
    float* __restrict__ C, int ldc, long sC,
    int M, int N, int K, int transB, int relu)
{
  int lane = threadIdx.x;
  int mL = lane & 15, g = lane >> 4;
  int m0 = blockIdx.x*16, n0 = blockIdx.y*16;
  const float* Arow = A + (long)blockIdx.z*sA + (m0 + mL)*lda;
  const float* Bbase = Bw + (long)blockIdx.z*sB;
  const float* Brow = transB ? (Bbase + (n0 + mL)*ldb) : (Bbase + (n0 + mL));
  int bk = transB ? 1 : ldb;           // k-stride for B gathers
  float* Cb = C + (long)blockIdx.z*sC;

  v8f acc = {};
  int kfull = K & ~31;
  for (int k0 = 0; k0 < kfull; k0 += 32){
    v16h a = load_tile_rowmajor(Arow, k0, g);
    v16h bm;
    if (transB){
      bm = load_tile_rowmajor(Brow, k0, g);
    } else {
#pragma unroll
      for (int h = 0; h < 16; ++h){
        int k = k0 + (h & 7) + 8*g + 16*(h >> 3);
        bm[h] = (_Float16)Brow[k*bk];
      }
    }
    acc = __builtin_amdgcn_wmma_f32_16x16x32_f16(false, a, false, bm, (short)0, acc, false, false);
  }
  if (kfull < K){                      // branchless clamped tail (e.g. K=16)
    v16h a, bm;
#pragma unroll
    for (int h = 0; h < 16; ++h){
      int k = kfull + (h & 7) + 8*g + 16*(h >> 3);
      int kc = (k < K) ? k : (K - 1);  // clamp: load stays in-bounds
      float av = Arow[kc];
      float bv = Brow[kc*bk];
      a[h]  = (_Float16)((k < K) ? av : 0.f);
      bm[h] = (_Float16)((k < K) ? bv : 0.f);
    }
    acc = __builtin_amdgcn_wmma_f32_16x16x32_f16(false, a, false, bm, (short)0, acc, false, false);
  }
  int n = n0 + mL;
  int mb = m0 + 8*g;
  float bv = bias ? bias[n] : 0.f;
#pragma unroll
  for (int r = 0; r < 8; ++r){
    float v = acc[r] + bv;
    if (relu) v = fmaxf(v, 0.f);
    Cb[(size_t)(mb+r)*ldc + n] = v;
  }
}

// =====================================================================
// Elementwise / small kernels
// =====================================================================
__global__ void bn_kernel(const float* __restrict__ x, const float* __restrict__ w,
                          const float* __restrict__ bb, const float* __restrict__ mean,
                          const float* __restrict__ var, float* __restrict__ y,
                          int D, int total)
{
  int i = blockIdx.x*blockDim.x + threadIdx.x;
  if (i >= total) return;
  int d = i % D;
  y[i] = (x[i] - mean[d]) * rsqrtf(var[d] + 1e-5f) * w[d] + bb[d];
}

__global__ void add_kernel(float* __restrict__ y, const float* __restrict__ a, int n)
{
  int i = blockIdx.x*blockDim.x + threadIdx.x;
  if (i < n) y[i] += a[i];
}

// GRU scan: one block per batch; 64 threads; xi (B,L,48) precomputed.
__global__ __launch_bounds__(64) void gru_scan_kernel(
    const float* __restrict__ xi, const float* __restrict__ Whh,
    const float* __restrict__ bhh, float* __restrict__ out,
    int reverse, int colOff)
{
  __shared__ float h[16];
  __shared__ float gh[48];
  int b = blockIdx.x, t = threadIdx.x;
  float w[16]; float bb = 0.f;
  if (t < 48){
    for (int k = 0; k < 16; ++k) w[k] = Whh[t*16 + k];
    bb = bhh[t];
  }
  if (t < 16) h[t] = 0.f;
  __syncthreads();
  for (int s = 0; s < LSEQ; ++s){
    int l = reverse ? (LSEQ-1-s) : s;
    const float* xr = xi + ((size_t)b*LSEQ + l)*48;
    if (t < 48){
      float acc = bb;
      for (int k = 0; k < 16; ++k) acc += w[k]*h[k];
      gh[t] = acc;
    }
    __syncthreads();
    if (t < 16){
      float r = sigmoidf_(xr[t]      + gh[t]);
      float z = sigmoidf_(xr[16 + t] + gh[16 + t]);
      float n = tanhf(xr[32 + t] + r*gh[32 + t]);
      float hn = (1.f - z)*n + z*h[t];
      out[((size_t)b*LSEQ + l)*32 + colOff + t] = hn;
      h[t] = hn;
    }
    __syncthreads();
  }
}

__global__ __launch_bounds__(256) void softmax_kernel(float* __restrict__ S, float scale)
{
  float* r = S + (size_t)blockIdx.x*LSEQ;
  int t = threadIdx.x;
  __shared__ float red[256];
  float m = -1e30f;
  for (int j = t; j < LSEQ; j += 256) m = fmaxf(m, r[j]*scale);
  red[t] = m; __syncthreads();
  for (int s = 128; s > 0; s >>= 1){ if (t < s) red[t] = fmaxf(red[t], red[t+s]); __syncthreads(); }
  m = red[0]; __syncthreads();
  float sum = 0.f;
  for (int j = t; j < LSEQ; j += 256){ float e = __expf(r[j]*scale - m); r[j] = e; sum += e; }
  red[t] = sum; __syncthreads();
  for (int s = 128; s > 0; s >>= 1){ if (t < s) red[t] += red[t+s]; __syncthreads(); }
  float inv = 1.f/red[0];
  for (int j = t; j < LSEQ; j += 256) r[j] *= inv;
}

__global__ void ln_res_kernel(float* __restrict__ x, const float* __restrict__ a,
                              const float* __restrict__ w, const float* __restrict__ bb,
                              int rows)
{
  int r = blockIdx.x*blockDim.x + threadIdx.x;
  if (r >= rows) return;
  float* xr = x + (size_t)r*32;
  const float* ar = a + (size_t)r*32;
  float v[32]; float mu = 0.f;
  for (int c = 0; c < 32; ++c){ v[c] = xr[c] + ar[c]; mu += v[c]; }
  mu *= (1.f/32.f);
  float var = 0.f;
  for (int c = 0; c < 32; ++c){ float d = v[c]-mu; var += d*d; }
  var *= (1.f/32.f);
  float inv = rsqrtf(var + 1e-5f);
  for (int c = 0; c < 32; ++c) xr[c] = (v[c]-mu)*inv*w[c] + bb[c];
}

// =====================================================================
// Pairwise projection: pf[b, i*L+j, c] = sum_e (x_i[e]*x_j[e]*scale)*W[c,e] + b[c]
// One block per (b,i); x_i row + convW staged in LDS via async loads.
// =====================================================================
__global__ __launch_bounds__(128) void pairproj_kernel(
    const float* __restrict__ X, const float* __restrict__ convW,
    const float* __restrict__ convB, const float* __restrict__ scale_p,
    float* __restrict__ pf)
{
  __shared__ float s_xi[32];
  __shared__ float s_w[1024];
  int blk = blockIdx.x;
  int b = blk >> 9, i = blk & (LSEQ-1);
  int t = threadIdx.x;
  if (t < 32) async_b32(&s_xi[t], &X[((size_t)b*LSEQ + i)*32 + t]);
  for (int q = t; q < 1024; q += 128) async_b32(&s_w[q], &convW[q]);
  wait_async0();
  __syncthreads();
  float scale = scale_p[0];
  int wave = t >> 5, lane = t & 31, mL = lane & 15, g = lane >> 4;
  float* dst = pf + ((size_t)b*NPAIR + (size_t)i*LSEQ)*32;
  // premultiplied x_i (per-lane registers, k runs are contiguous)
  for (int jt = wave; jt < 32; jt += 4){
    int j0 = jt*16;
    const float* xr = X + ((size_t)b*LSEQ + j0 + mL)*32;
    v16h a;
    {
      const float4* x0 = (const float4*)(xr + 8*g);
      const float4* x1 = (const float4*)(xr + 16 + 8*g);
      const float4* s0 = (const float4*)(&s_xi[8*g]);
      const float4* s1 = (const float4*)(&s_xi[16 + 8*g]);
      float4 xa = x0[0], xb = x0[1], xc = x1[0], xd = x1[1];
      float4 sa = s0[0], sb = s0[1], sc = s1[0], sd = s1[1];
      float4 pa, pb, pc, pd;
      pa.x = xa.x*sa.x*scale; pa.y = xa.y*sa.y*scale; pa.z = xa.z*sa.z*scale; pa.w = xa.w*sa.w*scale;
      pb.x = xb.x*sb.x*scale; pb.y = xb.y*sb.y*scale; pb.z = xb.z*sb.z*scale; pb.w = xb.w*sb.w*scale;
      pc.x = xc.x*sc.x*scale; pc.y = xc.y*sc.y*scale; pc.z = xc.z*sc.z*scale; pc.w = xc.w*sc.w*scale;
      pd.x = xd.x*sd.x*scale; pd.y = xd.y*sd.y*scale; pd.z = xd.z*sd.z*scale; pd.w = xd.w*sd.w*scale;
      pack8lo(a, pa, pb);
      pack8hi(a, pc, pd);
    }
#pragma unroll
    for (int nt = 0; nt < 2; ++nt){
      const float* wr = &s_w[(nt*16 + mL)*32];
      v16h bm = load_tile_rowmajor(wr, 0, g);
      v8f acc = {};
      acc = __builtin_amdgcn_wmma_f32_16x16x32_f16(false, a, false, bm, (short)0, acc, false, false);
      int n = nt*16 + mL;
      float bv = convB[n];
      int m0 = j0 + 8*g;
#pragma unroll
      for (int r = 0; r < 8; ++r) dst[(size_t)(m0+r)*32 + n] = acc[r] + bv;
    }
  }
}

// KV[d,e] = sum_n phi(pf[n,d]) * pf[n,e]  (WMMA reduction + f32 atomics)
__global__ __launch_bounds__(128) void pair_kv_kernel(
    const float* __restrict__ pf, float* __restrict__ KV, int rowsPerChunk)
{
  int b = blockIdx.y;
  const float* p = pf + (size_t)b*NPAIR*32;
  int t = threadIdx.x, wave = t >> 5, lane = t & 31;
  int dt = wave >> 1, et = wave & 1;
  int n0 = blockIdx.x*rowsPerChunk;
  int mL = lane & 15, g = lane >> 4;
  v8f acc = {};
  for (int n = n0; n < n0 + rowsPerChunk; n += 32){
    const float* pr = p + (size_t)n*32;
    __builtin_prefetch(pr + 32*32, 0, 1);     // global_prefetch: next 32-row block
    v16h a, bm;
#pragma unroll
    for (int h = 0; h < 16; ++h){
      int kk = (h & 7) + 8*g + 16*(h >> 3);
      a[h]  = (_Float16)phi_elu(pr[kk*32 + dt*16 + mL]);
      bm[h] = (_Float16)(pr[kk*32 + et*16 + mL]);
    }
    acc = __builtin_amdgcn_wmma_f32_16x16x32_f16(false, a, false, bm, (short)0, acc, false, false);
  }
  float* kv = KV + b*1024;
  int e = et*16 + mL, d0 = dt*16 + 8*g;
#pragma unroll
  for (int r = 0; r < 8; ++r) atomicAdd(&kv[(d0+r)*32 + e], acc[r]);
}

__global__ __launch_bounds__(256) void pair_ksum_kernel(
    const float* __restrict__ pf, float* __restrict__ Ksum, int rowsPerChunk)
{
  int b = blockIdx.y;
  const float* p = pf + (size_t)b*NPAIR*32;
  int t = threadIdx.x, d = t & 31, grp = t >> 5;
  int n0 = blockIdx.x*rowsPerChunk;
  float s = 0.f;
  for (int n = n0 + grp; n < n0 + rowsPerChunk; n += 8)
    s += phi_elu(p[(size_t)n*32 + d]);
  __shared__ float red[256];
  red[t] = s; __syncthreads();
  if (grp == 0){
    float a = 0.f;
    for (int g2 = 0; g2 < 8; ++g2) a += red[g2*32 + d];
    atomicAdd(&Ksum[b*32 + d], a);
  }
}

// fused: z, ctx = (phi(pf)@KV)/z, pf = LN(pf + ctx), in-place
__global__ void pair_ctx_ln_kernel(float* __restrict__ pf, const float* __restrict__ KV,
                                   const float* __restrict__ Ksum,
                                   const float* __restrict__ lnw, const float* __restrict__ lnb)
{
  size_t idx = (size_t)blockIdx.x*blockDim.x + threadIdx.x;
  int b = (int)(idx >> 18);
  float* p = pf + idx*32;
  float v[32], qp[32];
  float z = 1e-6f;
  for (int d = 0; d < 32; ++d){ v[d] = p[d]; qp[d] = phi_elu(v[d]); z += qp[d]*Ksum[b*32 + d]; }
  float ctx[32];
  for (int e = 0; e < 32; ++e) ctx[e] = 0.f;
  const float* kv = KV + b*1024;
  for (int d = 0; d < 32; ++d){
    float q = qp[d];
    for (int e = 0; e < 32; ++e) ctx[e] += q*kv[d*32 + e];
  }
  float invz = 1.f/z;
  float mu = 0.f;
  for (int e = 0; e < 32; ++e){ v[e] += ctx[e]*invz; mu += v[e]; }
  mu *= (1.f/32.f);
  float var = 0.f;
  for (int e = 0; e < 32; ++e){ float d0 = v[e]-mu; var += d0*d0; }
  var *= (1.f/32.f);
  float inv = rsqrtf(var + 1e-5f);
  for (int e = 0; e < 32; ++e) p[e] = (v[e]-mu)*inv*lnw[e] + lnb[e];
}

__global__ void sym_kernel(float* __restrict__ pw)
{
  size_t idx = (size_t)blockIdx.x*blockDim.x + threadIdx.x;
  int b = (int)(idx >> 18);
  int rem = (int)(idx & (NPAIR-1));
  int i = rem >> 9, j = rem & (LSEQ-1);
  if (i >= j) return;
  float* a = pw + ((size_t)b*NPAIR + (size_t)i*LSEQ + j)*32;
  float* c = pw + ((size_t)b*NPAIR + (size_t)j*LSEQ + i)*32;
#pragma unroll
  for (int k = 0; k < 32; ++k){ float m = 0.5f*(a[k] + c[k]); a[k] = m; c[k] = m; }
}

// =====================================================================
// Fused refinement: pw += relu(LN(pw)@W1^T + b1)@W2^T + b2, 16 rows/block
// =====================================================================
__global__ __launch_bounds__(128) void refine_kernel(
    float* __restrict__ pw,
    const float* __restrict__ lnw, const float* __restrict__ lnb,
    const float* __restrict__ w1, const float* __restrict__ b1,
    const float* __restrict__ w2, const float* __restrict__ b2)
{
  __shared__ float s_w1[64*32];
  __shared__ float s_w2[32*64];
  __shared__ float s_b1[64];
  __shared__ float s_b2[32];
  __shared__ float s_ln[64];
  __shared__ float s_pn[16*32];
  __shared__ float s_h[16*64];
  __shared__ float s_part[4][16*16];
  int t = threadIdx.x;
  for (int q = t; q < 2048; q += 128) async_b32(&s_w1[q], &w1[q]);
  for (int q = t; q < 2048; q += 128) async_b32(&s_w2[q], &w2[q]);
  if (t < 64) async_b32(&s_b1[t], &b1[t]);
  if (t < 32){
    async_b32(&s_b2[t], &b2[t]);
    async_b32(&s_ln[t], &lnw[t]);
    async_b32(&s_ln[32 + t], &lnb[t]);
  }
  wait_async0();
  __syncthreads();
  size_t row0 = (size_t)blockIdx.x*16;
  if (t < 16){
    const float* pr = pw + (row0 + t)*32;
    float v[32]; float mu = 0.f;
    for (int c = 0; c < 32; ++c){ v[c] = pr[c]; mu += v[c]; }
    mu *= (1.f/32.f);
    float var = 0.f;
    for (int c = 0; c < 32; ++c){ float d = v[c]-mu; var += d*d; }
    var *= (1.f/32.f);
    float inv = rsqrtf(var + 1e-5f);
    for (int c = 0; c < 32; ++c) s_pn[t*32 + c] = (v[c]-mu)*inv*s_ln[c] + s_ln[32 + c];
  }
  __syncthreads();
  int wave = t >> 5, lane = t & 31, mL = lane & 15, g = lane >> 4;
  { // GEMM1: h = relu(pn(16x32) @ w1^T(32x64)); wave owns 16 output cols
    v16h a = load_tile_rowmajor(&s_pn[mL*32], 0, g);
    v16h bm = load_tile_rowmajor(&s_w1[(wave*16 + mL)*32], 0, g);
    v8f acc = {};
    acc = __builtin_amdgcn_wmma_f32_16x16x32_f16(false, a, false, bm, (short)0, acc, false, false);
    int n = wave*16 + mL, m0 = 8*g;
#pragma unroll
    for (int r = 0; r < 8; ++r) s_h[(m0+r)*64 + n] = fmaxf(acc[r] + s_b1[n], 0.f);
  }
  __syncthreads();
  { // GEMM2: out = h(16x64) @ w2^T(64x32); K split over wave pairs
    int nt = wave & 1, ks = wave >> 1;
    v16h a = load_tile_rowmajor(&s_h[mL*64], ks*32, g);
    v16h bm = load_tile_rowmajor(&s_w2[(nt*16 + mL)*64], ks*32, g);
    v8f acc = {};
    acc = __builtin_amdgcn_wmma_f32_16x16x32_f16(false, a, false, bm, (short)0, acc, false, false);
    int m0 = 8*g;
#pragma unroll
    for (int r = 0; r < 8; ++r) s_part[wave][(m0+r)*16 + mL] = acc[r];
  }
  __syncthreads();
  for (int q = t; q < 512; q += 128){
    int m = q >> 5, c = q & 31;
    float v = s_part[c >> 4][m*16 + (c & 15)] + s_part[(c >> 4) + 2][m*16 + (c & 15)] + s_b2[c];
    pw[(row0 + m)*32 + c] += v;
  }
}

// =====================================================================
// Output regression -> dm (directly into d_out)
// =====================================================================
__global__ void outreg_kernel(const float* __restrict__ pw, const float* __restrict__ cw,
                              const float* __restrict__ cb, const float* __restrict__ gain,
                              float* __restrict__ dm)
{
  size_t idx = (size_t)blockIdx.x*blockDim.x + threadIdx.x;
  int rem = (int)(idx & (NPAIR-1));
  int i = rem >> 9, j = rem & (LSEQ-1);
  const float* p = pw + idx*32;
  float acc = cb[0];
  for (int c = 0; c < 32; ++c) acc += p[c]*cw[c];
  float tt = sigmoidf_(gain[0])*acc;
  float sp = (tt > 20.f) ? tt : log1pf(__expf(tt));
  sp = fminf(sp, 100.f);
  float s = sigmoidf_(sp);
  dm[idx] = (i == j) ? 0.f : s;
}

// =====================================================================
// MDS: double-centering via means, then top-3 subspace iteration
// =====================================================================
__global__ __launch_bounds__(256) void mds_rowstats_kernel(const float* __restrict__ dm,
                                                           float* __restrict__ rmean)
{
  int b = blockIdx.y, i = blockIdx.x, t = threadIdx.x;
  const float* r = dm + ((size_t)b*LSEQ + i)*LSEQ;
  __shared__ float red[256];
  float s = 0.f;
  for (int j = t; j < LSEQ; j += 256){
    float d = fminf(fmaxf(r[j], 0.f), 100.f);
    s += d*d + 1e-6f;
  }
  red[t] = s; __syncthreads();
  for (int st = 128; st > 0; st >>= 1){ if (t < st) red[t] += red[t+st]; __syncthreads(); }
  if (t == 0) rmean[b*LSEQ + i] = red[0]/(float)LSEQ;
}

__global__ __launch_bounds__(256) void mds_total_kernel(const float* __restrict__ rmean,
                                                        float* __restrict__ tmean)
{
  int b = blockIdx.x, t = threadIdx.x;
  __shared__ float red[256];
  float s = 0.f;
  for (int i = t; i < LSEQ; i += 256) s += rmean[b*LSEQ + i];
  red[t] = s; __syncthreads();
  for (int st = 128; st > 0; st >>= 1){ if (t < st) red[t] += red[t+st]; __syncthreads(); }
  if (t == 0) tmean[b] = red[0]/(float)LSEQ;
}

__global__ void mds_center_kernel(const float* __restrict__ dm, const float* __restrict__ rmean,
                                  const float* __restrict__ tmean, float* __restrict__ Bmat)
{
  size_t idx = (size_t)blockIdx.x*blockDim.x + threadIdx.x;
  int b = (int)(idx >> 18);
  int rem = (int)(idx & (NPAIR-1));
  int i = rem >> 9, j = rem & (LSEQ-1);
  float d = fminf(fmaxf(dm[idx], 0.f), 100.f);
  float d2 = d*d + 1e-6f;
  float v = -0.5f*(d2 - rmean[b*LSEQ + i] - rmean[b*LSEQ + j] + tmean[b]);
  if (i == j) v += 1e-3f;
  Bmat[idx] = v;
}

__device__ inline float block_dot512(const float* a, const float* bv, int t, float* red){
  red[t] = a[t]*bv[t]; __syncthreads();
  for (int s = 256; s > 0; s >>= 1){ if (t < s) red[t] += red[t+s]; __syncthreads(); }
  float r = red[0]; __syncthreads();
  return r;
}

__global__ __launch_bounds__(512) void mds_topk_kernel(const float* __restrict__ Bmat,
                                                       const float* __restrict__ mask,
                                                       float* __restrict__ coords)
{
  __shared__ float Q[3][LSEQ];
  __shared__ float Y[3][LSEQ];
  __shared__ float red[LSEQ];
  __shared__ float lam[3];
  int b = blockIdx.x, t = threadIdx.x;
  const float* A = Bmat + (size_t)b*NPAIR;
  for (int k = 0; k < 3; ++k) Q[k][t] = sinf(0.37f*(float)((t+1)*(k+1)) + 0.11f*(float)k);
  __syncthreads();
  for (int it = 0; it < 60; ++it){
    const float* row = A + (size_t)t*LSEQ;
    for (int k = 0; k < 3; ++k){
      float s = 0.f;
      for (int j = 0; j < LSEQ; ++j) s += row[j]*Q[k][j];
      Y[k][t] = s;
    }
    __syncthreads();
    for (int k = 0; k < 3; ++k){
      for (int p = 0; p < k; ++p){
        float d = block_dot512(Q[p], Y[k], t, red);
        Y[k][t] -= d*Q[p][t];
      }
      float nn = block_dot512(Y[k], Y[k], t, red);
      Q[k][t] = Y[k][t]*rsqrtf(fmaxf(nn, 1e-30f));
      __syncthreads();
    }
  }
  {
    const float* row = A + (size_t)t*LSEQ;
    for (int k = 0; k < 3; ++k){
      float s = 0.f;
      for (int j = 0; j < LSEQ; ++j) s += row[j]*Q[k][j];
      Y[k][t] = s;
    }
  }
  __syncthreads();
  for (int k = 0; k < 3; ++k){
    float l = block_dot512(Q[k], Y[k], t, red);
    if (t == 0) lam[k] = l;
  }
  __syncthreads();
  float mk = mask[b*LSEQ + t];
  for (int c = 0; c < 3; ++c){
    int k = 2 - c;                               // ascending order, q0 = dominant
    float lv = lam[k] > 0.f ? sqrtf(lam[k]) : 0.f;
    coords[((size_t)b*LSEQ + t)*3 + c] = Q[k][t]*lv*mk;
  }
}

// =====================================================================
// Host orchestration
// =====================================================================
extern "C" void kernel_launch(void* const* d_in, const int* in_sizes, int n_in,
                              void* d_out, int out_size, void* d_ws, size_t ws_size,
                              hipStream_t stream)
{
  (void)in_sizes; (void)n_in; (void)out_size; (void)ws_size;
  auto F = [&](int i){ return (const float*)d_in[i]; };
  float* ws = (float*)d_ws;
  size_t off = 0;
  auto alloc = [&](size_t n){ float* p = ws + off; off += (n + 63) & ~(size_t)63; return p; };

  float* pf     = alloc((size_t)NB*NPAIR*32);
  float* scores = alloc((size_t)NB*NPAIR);
  float* Bmat   = alloc((size_t)NB*NPAIR);
  float* xn     = alloc((size_t)NB*LSEQ*128);
  float* xiF    = alloc((size_t)NB*LSEQ*48);
  float* xiB    = alloc((size_t)NB*LSEQ*48);
  float* xA     = alloc((size_t)NB*LSEQ*32);
  float* xB     = alloc((size_t)NB*LSEQ*32);
  float* xCv    = alloc((size_t)NB*LSEQ*32);
  float* resb   = alloc((size_t)NB*LSEQ*32);
  float* t0     = alloc((size_t)NB*LSEQ*96);
  float* t1     = alloc((size_t)NB*LSEQ*32);
  float* t2     = alloc((size_t)NB*LSEQ*32);
  float* KV     = alloc((size_t)NB*1024);
  float* Ksum   = alloc((size_t)NB*32);
  float* rmean  = alloc((size_t)NB*LSEQ);
  float* tmean  = alloc((size_t)NB);

  auto GEMM = [&](const float* A, int lda, long sA, const float* Bw, int ldb, long sB,
                  const float* bias, float* C, int ldc, long sC,
                  int M, int N, int K, int transB, int relu, int batch){
    dim3 g(M/16, N/16, batch);
    gemm_wmma_kernel<<<g, 32, 0, stream>>>(A, lda, sA, Bw, ldb, sB, bias, C, ldc, sC,
                                           M, N, K, transB, relu);
  };

  const int ROWS = NB*LSEQ;   // 1024

  // ---- RecurrentBackbone ----
  float* gb[3] = {xA, xB, xCv};
  int base[3] = {2, 16, 28};
  for (int li = 0; li < 3; ++li){
    int bs = base[li];
    const float* src = (li == 0) ? F(0) : gb[li-1];
    int D = (li == 0) ? 128 : 32;
    int tot = ROWS*D;
    bn_kernel<<<(tot+255)/256, 256, 0, stream>>>(src, F(bs+0), F(bs+1), F(bs+2), F(bs+3), xn, D, tot);
    GEMM(xn, D, 0, F(bs+4), D, 0, F(bs+6), xiF, 48, 0, ROWS, 48, D, 1, 0, 1);
    GEMM(xn, D, 0, F(bs+8), D, 0, F(bs+10), xiB, 48, 0, ROWS, 48, D, 1, 0, 1);
    gru_scan_kernel<<<NB, 64, 0, stream>>>(xiF, F(bs+5), F(bs+7), gb[li], 0, 0);
    gru_scan_kernel<<<NB, 64, 0, stream>>>(xiB, F(bs+9), F(bs+11), gb[li], 1, 16);
    if (li == 0){
      GEMM(F(0), 128, 0, F(bs+12), 128, 0, F(bs+13), resb, 32, 0, ROWS, 32, 128, 1, 0, 1);
      add_kernel<<<(ROWS*32+255)/256, 256, 0, stream>>>(gb[0], resb, ROWS*32);
    } else {
      add_kernel<<<(ROWS*32+255)/256, 256, 0, stream>>>(gb[li], src, ROWS*32);
    }
  }
  float* X = xCv;

  // ---- LowRankSelfAttention ----
  float* Qb = t0;
  float* Kb = t0 + ROWS*16;
  float* Vb = t0 + 2*ROWS*16;
  GEMM(X, 32, 0, F(40), 32, 0, F(41), Qb, 16, 0, ROWS, 16, 32, 1, 0, 1);
  GEMM(X, 32, 0, F(42), 32, 0, F(43), Kb, 16, 0, ROWS, 16, 32, 1, 0, 1);
  GEMM(X, 32, 0, F(44), 32, 0, F(45), Vb, 16, 0, ROWS, 16, 32, 1, 0, 1);
  GEMM(Qb, 16, (long)LSEQ*16, Kb, 16, (long)LSEQ*16, nullptr,
       scores, LSEQ, (long)NPAIR, LSEQ, LSEQ, 16, 1, 0, NB);
  softmax_kernel<<<ROWS, 256, 0, stream>>>(scores, 0.25f);
  GEMM(scores, LSEQ, (long)NPAIR, Vb, 16, (long)LSEQ*16, nullptr,
       t1, 16, (long)LSEQ*16, LSEQ, 16, LSEQ, 0, 0, NB);
  GEMM(t1, 16, 0, F(46), 16, 0, F(47), t2, 32, 0, ROWS, 32, 16, 1, 0, 1);
  add_kernel<<<(ROWS*32+255)/256, 256, 0, stream>>>(X, t2, ROWS*32);

  // ---- MiniTransformerBlock ----
  GEMM(X, 32, 0, F(48), 32, 0, F(49), t0, 96, 0, ROWS, 96, 32, 1, 0, 1);
  GEMM(t0, 96, (long)LSEQ*96, t0 + 32, 96, (long)LSEQ*96, nullptr,
       scores, LSEQ, (long)NPAIR, LSEQ, LSEQ, 32, 1, 0, NB);
  softmax_kernel<<<ROWS, 256, 0, stream>>>(scores, 0.17677669529663687f);
  GEMM(scores, LSEQ, (long)NPAIR, t0 + 64, 96, (long)LSEQ*96, nullptr,
       t1, 32, (long)LSEQ*32, LSEQ, 32, LSEQ, 0, 0, NB);
  GEMM(t1, 32, 0, F(50), 32, 0, F(51), t2, 32, 0, ROWS, 32, 32, 1, 0, 1);
  ln_res_kernel<<<(ROWS+255)/256, 256, 0, stream>>>(X, t2, F(52), F(53), ROWS);
  GEMM(X, 32, 0, F(54), 32, 0, F(55), t1, 32, 0, ROWS, 32, 32, 1, 1, 1);
  GEMM(t1, 32, 0, F(56), 32, 0, F(57), t2, 32, 0, ROWS, 32, 32, 1, 0, 1);
  ln_res_kernel<<<(ROWS+255)/256, 256, 0, stream>>>(X, t2, F(58), F(59), ROWS);

  // ---- PairwiseProjection + linear attention + LN + symmetrize ----
  pairproj_kernel<<<ROWS, 128, 0, stream>>>(X, F(61), F(62), F(60), pf);
  hipMemsetAsync(KV, 0, (size_t)NB*1024*sizeof(float), stream);
  hipMemsetAsync(Ksum, 0, (size_t)NB*32*sizeof(float), stream);
  pair_kv_kernel<<<dim3(128, NB), 128, 0, stream>>>(pf, KV, NPAIR/128);
  pair_ksum_kernel<<<dim3(128, NB), 256, 0, stream>>>(pf, Ksum, NPAIR/128);
  pair_ctx_ln_kernel<<<(NB*NPAIR)/256, 256, 0, stream>>>(pf, KV, Ksum, F(63), F(64));
  sym_kernel<<<(NB*NPAIR)/256, 256, 0, stream>>>(pf);

  // ---- IterativeRefinement ----
  refine_kernel<<<(NB*NPAIR)/16, 128, 0, stream>>>(pf, F(65), F(66), F(67), F(68), F(69), F(70));
  sym_kernel<<<(NB*NPAIR)/256, 256, 0, stream>>>(pf);

  // ---- OutputRegression -> dm in d_out ----
  float* dm = (float*)d_out;
  outreg_kernel<<<(NB*NPAIR)/256, 256, 0, stream>>>(pf, F(71), F(72), F(73), dm);

  // ---- Differentiable MDS ----
  mds_rowstats_kernel<<<dim3(LSEQ, NB), 256, 0, stream>>>(dm, rmean);
  mds_total_kernel<<<NB, 256, 0, stream>>>(rmean, tmean);
  mds_center_kernel<<<(NB*NPAIR)/256, 256, 0, stream>>>(dm, rmean, tmean, Bmat);
  mds_topk_kernel<<<NB, 512, 0, stream>>>(Bmat, F(1), dm + (size_t)NB*NPAIR);
}
